// GCM_60490319396973
// MI455X (gfx1250) — compile-verified, matrix operand order checked by reading, not run on hardware
//
#include <hip/hip_runtime.h>
#include <math.h>

typedef __bf16 bf16;
typedef __attribute__((ext_vector_type(16))) __bf16 v16bf;
typedef __attribute__((ext_vector_type(8)))  __bf16 v8bf;
typedef __attribute__((ext_vector_type(8)))  float  v8f;

#define DIMC 256
#define HC 128
#define KC 27
#define BC 4

// ---------------- WMMA helper ----------------

__device__ __forceinline__ v8f wmma_bf16(v16bf a, v16bf b, v8f c) {
  return __builtin_amdgcn_wmma_f32_16x16x32_bf16(false, a, false, b, (short)0, c, false, false);
}

// Swizzled fragment layout (per 32-k chunk):
//   section = [subtile or rowblock t][lane 0..31][16 halves], 16 halves contiguous (32B).
//   For element (k within chunk, n): lane = ((k>>3)&1)*16 + (n&15), i = (k&7) + ((k>>4)<<3).
// Fragment load = one aligned 32B LDS read.

// ---------------- small utility kernels ----------------

// f32 -> bf16 with WMMA-fragment swizzle. Matrices of KT x NT, nmat of them.
__global__ void convert_swz_kernel(const float* __restrict__ src, bf16* __restrict__ dst,
                                   int KT, int NT, int total) {
  int i = blockIdx.x * blockDim.x + threadIdx.x;
  if (i >= total) return;
  int per = KT * NT;
  int mat = i / per;
  int rem = i - mat * per;
  int k = rem / NT, n = rem - (rem / NT) * NT;
  int kc = k >> 5, kk = k & 31;
  int lane = ((kk >> 3) & 1) * 16 + (n & 15);
  int ii = (kk & 7) + ((kk >> 4) << 3);
  size_t o = (size_t)mat * per + (size_t)kc * 32 * NT + ((size_t)(n >> 4) * 32 + lane) * 16 + ii;
  dst[o] = (bf16)src[i];
}

__global__ void zero_f32_kernel(float* __restrict__ p, int n) {
  int i = blockIdx.x * blockDim.x + threadIdx.x;
  if (i < n) p[i] = 0.0f;
}

// cg[:,0:128] = r2 + 2*cg[:,0:128], 8 elements per thread (vectorized)
__global__ void combine_conv_kernel(const bf16* __restrict__ r2, bf16* __restrict__ cg, int total8) {
  int i = blockIdx.x * blockDim.x + threadIdx.x;
  if (i >= total8) return;
  int e = i * 8;
  int n = e >> 7, j = e & 127;
  v8bf a = *(const v8bf*)(r2 + e);
  v8bf c = *(const v8bf*)(cg + (size_t)n * DIMC + j);
  v8bf o;
#pragma unroll
  for (int u = 0; u < 8; ++u) o[u] = (bf16)((float)a[u] + 2.0f * (float)c[u]);
  *(v8bf*)(cg + (size_t)n * DIMC + j) = o;
}

// segment-sum of f2 over batches + counts. 128 threads/block, 256 rows/block.
__global__ void reduce_m2_kernel(const bf16* __restrict__ f2, const int* __restrict__ bids,
                                 float* __restrict__ m2, float* __restrict__ counts, int nrows) {
  int j  = threadIdx.x;
  int r0 = blockIdx.x * 256;
  float acc[BC] = {0.f, 0.f, 0.f, 0.f};
  int   cnt[BC] = {0, 0, 0, 0};
  for (int r = 0; r < 256; ++r) {
    int node = r0 + r;
    if (node >= nrows) break;
    int b = bids[node] & (BC - 1);
    acc[b] += (float)f2[(size_t)node * HC + j];
    if (j == 0) cnt[b]++;
  }
#pragma unroll
  for (int b = 0; b < BC; ++b) {
    atomicAdd(&m2[b * HC + j], acc[b]);
    if (j == 0 && cnt[b]) atomicAdd(&counts[b], (float)cnt[b]);
  }
}

// t = sqrt(mean(f1,-1) * m2[batch]/count + 1e-12) + f1 + f2  (bf16 out)
__global__ __launch_bounds__(256) void build_t_kernel(
    const bf16* __restrict__ f1, const bf16* __restrict__ f2,
    const int* __restrict__ bids, const float* __restrict__ m2,
    const float* __restrict__ counts, bf16* __restrict__ T, int nrows) {
  __shared__ float sPart[64 * 4];
  __shared__ float sRow[64];
  __shared__ float sM2[BC * HC];
  int tid = threadIdx.x;
  int row0 = blockIdx.x * 64;
  for (int i = tid; i < BC * HC; i += 256) {
    float c = counts[i >> 7];
    sM2[i] = m2[i] / fmaxf(c, 1.0f);
  }
  {
    int r = tid >> 2, q = tid & 3;
    int n = row0 + r;
    float s = 0.f;
    if (n < nrows) {
      const v8bf* p = (const v8bf*)(f1 + (size_t)n * HC + q * 32);
#pragma unroll
      for (int u = 0; u < 4; ++u) {
        v8bf t = p[u];
#pragma unroll
        for (int c = 0; c < 8; ++c) s += (float)t[c];
      }
    }
    sPart[r * 4 + q] = s;
  }
  __syncthreads();
  if (tid < 64)
    sRow[tid] = (sPart[tid * 4] + sPart[tid * 4 + 1] + sPart[tid * 4 + 2] + sPart[tid * 4 + 3]) * (1.0f / 128.0f);
  __syncthreads();
  for (int i = tid; i < 64 * HC; i += 256) {
    int r = i >> 7, j = i & 127;
    int n = row0 + r;
    if (n < nrows) {
      int b = bids[n] & (BC - 1);
      float enc = sqrtf(sRow[r] * sM2[b * HC + j] + 1e-12f);
      float v = enc + (float)f1[(size_t)n * HC + j] + (float)f2[(size_t)n * HC + j];
      T[(size_t)n * HC + j] = (bf16)v;
    }
  }
}

// ---------------- dense GEMM (B pre-swizzled) ----------------
// EPI: 0=store bf16, 2=glo: bf16(relu(aux - relu(v))), 3=f32 xres + v
template <int NT, int KT, bool AF32, int EPI>
__global__ __launch_bounds__(256) void pconv_kernel(
    const void* __restrict__ Av, int lda,
    const bf16* __restrict__ B, const float* __restrict__ bias,
    const bf16* __restrict__ aux, int ldaux, int auxcol,
    const float* __restrict__ xres,
    bf16* __restrict__ Ob, int ldo, int ocol,
    float* __restrict__ Of, int nrows) {
  constexpr int NSUB = NT / 32;  // 16-wide subtiles per wave (2 waves across N)
  __shared__ __align__(32) bf16 sA[64 * 32];
  __shared__ __align__(32) bf16 sB[32 * NT];
  int tid = threadIdx.x, lane = tid & 31, w = tid >> 5;
  int wm = w & 3, wn = w >> 2;
  int row0 = blockIdx.x * 64;
  v8f acc[NSUB];
#pragma unroll
  for (int s = 0; s < NSUB; ++s) acc[s] = (v8f){0.f, 0.f, 0.f, 0.f, 0.f, 0.f, 0.f, 0.f};

  for (int kc = 0; kc < KT; kc += 32) {
    {  // stage A: one 16B chunk per thread (row r, k-octet o)
      int r = tid >> 2, o = tid & 3;
      int n = row0 + r;
      int dst = (((r >> 4) * 32 + (o & 1) * 16 + (r & 15)) << 4) + ((o >> 1) << 3);
      if (AF32) {
        float4 lo = {0.f, 0.f, 0.f, 0.f}, hi = {0.f, 0.f, 0.f, 0.f};
        if (n < nrows) {
          const float* ap = (const float*)Av + (size_t)n * lda + kc + o * 8;
          lo = *(const float4*)ap;
          hi = *(const float4*)(ap + 4);
        }
        v8bf t;
        t[0] = (bf16)lo.x; t[1] = (bf16)lo.y; t[2] = (bf16)lo.z; t[3] = (bf16)lo.w;
        t[4] = (bf16)hi.x; t[5] = (bf16)hi.y; t[6] = (bf16)hi.z; t[7] = (bf16)hi.w;
        *(v8bf*)(sA + dst) = t;
      } else {
        uint4 v = {0u, 0u, 0u, 0u};
        if (n < nrows) v = *(const uint4*)((const bf16*)Av + (size_t)n * lda + kc + o * 8);
        *(uint4*)(sA + dst) = v;
      }
    }
    {  // stage B chunk: straight vector copy of pre-swizzled weights
      const uint4* bs = (const uint4*)(B + (size_t)(kc >> 5) * 32 * NT);
      uint4* bd = (uint4*)sB;
#pragma unroll
      for (int i = tid; i < 4 * NT; i += 256) bd[i] = bs[i];
    }
    __syncthreads();
    v16bf a = *(const v16bf*)(sA + ((wm * 32 + lane) << 4));
#pragma unroll
    for (int s = 0; s < NSUB; ++s) {
      v16bf bfr = *(const v16bf*)(sB + (((wn * NSUB + s) * 32 + lane) << 4));
      acc[s] = wmma_bf16(a, bfr, acc[s]);
    }
    __syncthreads();
  }
#pragma unroll
  for (int s = 0; s < NSUB; ++s) {
    int j = wn * (NT / 2) + s * 16 + (lane & 15);
#pragma unroll
    for (int r = 0; r < 8; ++r) {
      int n = row0 + wm * 16 + r + ((lane >> 4) << 3);
      if (n >= nrows) continue;
      float v = acc[s][r] + bias[j];
      if (EPI == 0) {
        Ob[(size_t)n * ldo + ocol + j] = (bf16)v;
      } else if (EPI == 2) {
        float o = fmaxf(v, 0.f);
        float g = (float)aux[(size_t)n * ldaux + auxcol + j];
        Ob[(size_t)n * ldo + ocol + j] = (bf16)fmaxf(g - o, 0.f);
      } else {
        Of[(size_t)n * ldo + j] = xres[(size_t)n * ldo + j] + v;
      }
    }
  }
}

// ---------------- gathered sconv GEMM (full k-slab staging) ----------------
// out[n,:] = relu(bias + sum_k feats[nbr[n,k]] @ W[k]) ; idx==nfeat -> zero row
template <bool DUAL>
__global__ __launch_bounds__(256) void sconv_kernel(
    const bf16* __restrict__ feats, int ldf, int fcol, int nfeat,
    const int* __restrict__ nbr,
    const bf16* __restrict__ Bw, const bf16* __restrict__ Bw2,
    const float* __restrict__ bias, const float* __restrict__ bias2,
    bf16* __restrict__ O1, bf16* __restrict__ O2, int ldo, int nrows) {
  __shared__ __align__(32) bf16 sA[4 * 2048];                 // 16 KB, 4 k-chunks
  __shared__ __align__(32) bf16 sB[4 * 4096];                 // 32 KB, full W[k]
  __shared__ __align__(32) bf16 sB2[DUAL ? 4 * 4096 : 16];
  __shared__ int sIdx[64];
  int tid = threadIdx.x, lane = tid & 31, w = tid >> 5;
  int wm = w & 3, wn = w >> 2;
  int row0 = blockIdx.x * 64;
  v8f acc[4], acc2[4];
#pragma unroll
  for (int s = 0; s < 4; ++s) {
    acc[s]  = (v8f){0.f, 0.f, 0.f, 0.f, 0.f, 0.f, 0.f, 0.f};
    acc2[s] = (v8f){0.f, 0.f, 0.f, 0.f, 0.f, 0.f, 0.f, 0.f};
  }
  for (int k = 0; k < KC; ++k) {
    __syncthreads();
    if (tid < 64) {
      int n = row0 + tid;
      sIdx[tid] = (n < nrows) ? nbr[(size_t)n * KC + k] : nfeat;
    }
    __syncthreads();
    // stage A: 64 rows x 128 cols, 16B chunks; 4 per thread
#pragma unroll
    for (int t4 = 0; t4 < 4; ++t4) {
      int lin = tid + t4 * 256;
      int r = lin >> 4, o = lin & 15;      // o = k-octet within 128
      int idx = sIdx[r];
      uint4 v = {0u, 0u, 0u, 0u};
      if ((unsigned)idx < (unsigned)nfeat)
        v = *(const uint4*)(feats + (size_t)idx * ldf + fcol + o * 8);
      int c = o >> 2, oo = o & 3;
      int dst = c * 2048 + ((((r >> 4) * 32 + (oo & 1) * 16 + (r & 15)) << 4) + ((oo >> 1) << 3));
      *(uint4*)(sA + dst) = v;
    }
    // stage full W[k] (pre-swizzled): 2048 uint4
    {
      const uint4* bs = (const uint4*)(Bw + (size_t)k * HC * HC);
      uint4* bd = (uint4*)sB;
#pragma unroll
      for (int t8 = 0; t8 < 8; ++t8) bd[tid + t8 * 256] = bs[tid + t8 * 256];
      if (DUAL) {
        const uint4* bs2 = (const uint4*)(Bw2 + (size_t)k * HC * HC);
        uint4* bd2 = (uint4*)sB2;
#pragma unroll
        for (int t8 = 0; t8 < 8; ++t8) bd2[tid + t8 * 256] = bs2[tid + t8 * 256];
      }
    }
    __syncthreads();
#pragma unroll
    for (int c = 0; c < 4; ++c) {
      v16bf a = *(const v16bf*)(sA + c * 2048 + ((wm * 32 + lane) << 4));
#pragma unroll
      for (int s = 0; s < 4; ++s) {
        v16bf bfr = *(const v16bf*)(sB + c * 4096 + (((wn * 4 + s) * 32 + lane) << 4));
        acc[s] = wmma_bf16(a, bfr, acc[s]);
        if (DUAL) {
          v16bf bfr2 = *(const v16bf*)(sB2 + c * 4096 + (((wn * 4 + s) * 32 + lane) << 4));
          acc2[s] = wmma_bf16(a, bfr2, acc2[s]);
        }
      }
    }
  }
#pragma unroll
  for (int s = 0; s < 4; ++s) {
    int j = wn * 64 + s * 16 + (lane & 15);
#pragma unroll
    for (int r = 0; r < 8; ++r) {
      int n = row0 + wm * 16 + r + ((lane >> 4) << 3);
      if (n >= nrows) continue;
      O1[(size_t)n * ldo + j] = (bf16)fmaxf(acc[s][r] + bias[j], 0.f);
      if (DUAL)
        O2[(size_t)n * ldo + j] = (bf16)fmaxf(acc2[s][r] + bias2[j], 0.f);
    }
  }
}

// ---------------- launcher ----------------

extern "C" void kernel_launch(void* const* d_in, const int* in_sizes, int n_in,
                              void* d_out, int out_size, void* d_ws, size_t ws_size,
                              hipStream_t stream) {
  const float* x    = (const float*)d_in[0];
  const int*   nbr  = (const int*)d_in[1];
  const int*   bids = (const int*)d_in[2];
  const float* W1   = (const float*)d_in[3];
  const float* b1   = (const float*)d_in[4];
  const float* W2   = (const float*)d_in[5];
  const float* b2   = (const float*)d_in[6];
  const float* Wr1  = (const float*)d_in[7];
  const float* br1  = (const float*)d_in[8];
  const float* Wr2  = (const float*)d_in[9];
  const float* br2  = (const float*)d_in[10];
  const float* Wg1  = (const float*)d_in[11];
  const float* bg1  = (const float*)d_in[12];
  const float* Wg2  = (const float*)d_in[13];
  const float* bg2  = (const float*)d_in[14];
  const float* Wg3  = (const float*)d_in[15];
  const float* bg3  = (const float*)d_in[16];
  float* out = (float*)d_out;

  const int N = in_sizes[0] / DIMC;

  // workspace carve-up
  char* p = (char*)d_ws;
  bf16* cg  = (bf16*)p; p += (size_t)N * DIMC * sizeof(bf16);   // h: conv|glo, updated in place
  bf16* r1  = (bf16*)p; p += (size_t)N * HC * sizeof(bf16);     // r1, reused as T
  bf16* r2  = (bf16*)p; p += (size_t)N * HC * sizeof(bf16);
  bf16* f1  = (bf16*)p; p += (size_t)N * HC * sizeof(bf16);
  bf16* f2  = (bf16*)p; p += (size_t)N * HC * sizeof(bf16);
  bf16* W1b  = (bf16*)p; p += (size_t)DIMC * DIMC * sizeof(bf16);
  bf16* W2b  = (bf16*)p; p += (size_t)DIMC * DIMC * sizeof(bf16);
  bf16* Wr1b = (bf16*)p; p += (size_t)KC * HC * HC * sizeof(bf16);
  bf16* Wr2b = (bf16*)p; p += (size_t)KC * HC * HC * sizeof(bf16);
  bf16* Wg1b = (bf16*)p; p += (size_t)KC * HC * HC * sizeof(bf16);
  bf16* Wg2b = (bf16*)p; p += (size_t)KC * HC * HC * sizeof(bf16);
  bf16* Wg3b = (bf16*)p; p += (size_t)HC * HC * sizeof(bf16);
  float* m2     = (float*)p; p += BC * HC * sizeof(float);
  float* counts = (float*)p; p += BC * sizeof(float);

  const int NB64 = (N + 63) / 64;

  // 1. weights -> bf16, pre-swizzled into WMMA fragment order
  auto cv = [&](const float* s, bf16* d, int KT, int NT, int nmat) {
    int tot = KT * NT * nmat;
    convert_swz_kernel<<<(tot + 255) / 256, 256, 0, stream>>>(s, d, KT, NT, tot);
  };
  cv(W1, W1b, DIMC, DIMC, 1);
  cv(W2, W2b, DIMC, DIMC, 1);
  cv(Wr1, Wr1b, HC, HC, KC);
  cv(Wr2, Wr2b, HC, HC, KC);
  cv(Wg1, Wg1b, HC, HC, KC);
  cv(Wg2, Wg2b, HC, HC, KC);
  cv(Wg3, Wg3b, HC, HC, 1);

  // 2. zero segment accumulators (graph replays)
  zero_f32_kernel<<<2, 256, 0, stream>>>(m2, BC * HC + BC);

  // 3. h = x@W1 + b1 -> cg (bf16, [N,256])
  pconv_kernel<DIMC, DIMC, true, 0><<<NB64, 256, 0, stream>>>(
      x, DIMC, W1b, b1, nullptr, 0, 0, nullptr, cg, DIMC, 0, nullptr, N);

  // 4. r1 = relu(sconv(conv_x, Wr1)+br1)
  sconv_kernel<false><<<NB64, 256, 0, stream>>>(
      cg, DIMC, 0, N, nbr, Wr1b, nullptr, br1, nullptr, r1, nullptr, HC, N);

  // 5. r2 = relu(sconv(r1, Wr2)+br2)
  sconv_kernel<false><<<NB64, 256, 0, stream>>>(
      r1, HC, 0, N, nbr, Wr2b, nullptr, br2, nullptr, r2, nullptr, HC, N);

  // 6. f1,f2 = relu(sconv(glo_x, Wg1/Wg2)) — shared gather
  sconv_kernel<true><<<NB64, 256, 0, stream>>>(
      cg, DIMC, HC, N, nbr, Wg1b, Wg2b, bg1, bg2, f1, f2, HC, N);

  // 7. m2/counts = segment sums of f2
  reduce_m2_kernel<<<(N + 255) / 256, 128, 0, stream>>>(f2, bids, m2, counts, N);

  // 8. T = enc + f1 + f2 (reuse r1 buffer)
  build_t_kernel<<<NB64, 256, 0, stream>>>(f1, f2, bids, m2, counts, r1, N);

  // 9. cg[:,128:] = relu(glo_x - relu(T@Wg3+bg3))  (in place)
  pconv_kernel<HC, HC, false, 2><<<NB64, 256, 0, stream>>>(
      r1, HC, Wg3b, bg3, cg, DIMC, HC, nullptr, cg, DIMC, HC, nullptr, N);

  // 10. cg[:,0:128] = r2 + 2*conv_x (in place, vectorized x8)
  combine_conv_kernel<<<(N * HC / 8 + 255) / 256, 256, 0, stream>>>(r2, cg, N * HC / 8);

  // 11. out = x + cg@W2 + b2 (f32)
  pconv_kernel<DIMC, DIMC, false, 3><<<NB64, 256, 0, stream>>>(
      cg, DIMC, W2b, b2, nullptr, 0, 0, x, nullptr, DIMC, 0, out, N);
}
